// DeepBSDE_1D_80925773791655
// MI455X (gfx1250) — matrix operand order repaired and checked by reading, not compiled, not run on hardware
//
#include <hip/hip_runtime.h>
#include <hip/hip_bf16.h>
#include <stdint.h>

// ---------------- problem constants (from reference) ----------------
typedef _Float16 v16h __attribute__((ext_vector_type(16)));
typedef float    v8f  __attribute__((ext_vector_type(8)));

constexpr int   NSTEPS = 64;
constexpr int   BATCH  = 524288;
constexpr int   HID    = 64;
constexpr float T_F    = 1.0f;
constexpr float R_F    = 0.05f;
constexpr float SIG    = 0.2f;
constexpr float KSTR   = 100.0f;
constexpr float DT     = T_F / NSTEPS;          // 1/64
constexpr float SQRT_DT = 0.125f;               // sqrt(1/64) exact
constexpr float DRIFT  = (R_F - 0.5f * SIG * SIG) * DT;

// per-step pre-swizzled weight blob layout in d_ws (16 KB each, 63 blobs)
constexpr int STEP_BYTES = 16384;
constexpr int W1S_OFF = 0;      // [2 kc][32 lane][16 i] f16 (A-layout features)
constexpr int B1S_OFF = 2048;   // same layout
constexpr int W2S_OFF = 4096;   // [8 tile][32 lane][16 i] f16 (B-layout tiles)
constexpr int W3F_OFF = 12288;  // [64] f32
constexpr int B2F_OFF = 12544;  // [64] f32
constexpr int B3F_OFF = 12800;  // [1]  f32

// A-operand (16-bit, 16x32) K index for v16h element i on a given lane
__device__ __host__ __forceinline__ int kmapA(int lane, int i) {
    return (i < 8 ? i : i + 8) + ((lane & 16) ? 8 : 0);
}

// single-instruction relu (compiler otherwise emits canonicalize-max + max)
__device__ __forceinline__ float relu1(float x) {
    float y;
    asm("v_max_num_f32 %0, 0, %1" : "=v"(y) : "v"(x));
    return y;
}

// ---------------- prep kernel: swizzle weights into WMMA layouts ----------------
__global__ void bsde_prep_kernel(const float* __restrict__ W1, const float* __restrict__ b1,
                                 const float* __restrict__ W2, const float* __restrict__ b2,
                                 const float* __restrict__ W3, const float* __restrict__ b3,
                                 uint8_t* __restrict__ ws)
{
    const int s = blockIdx.x;          // 0..62  (weights for sim step s+1)
    const int t = threadIdx.x;         // 0..255
    uint8_t* blob = ws + (size_t)s * STEP_BYTES;
    _Float16* w1s = (_Float16*)(blob + W1S_OFF);
    _Float16* b1s = (_Float16*)(blob + B1S_OFF);
    _Float16* w2s = (_Float16*)(blob + W2S_OFF);
    float*    w3f = (float*)(blob + W3F_OFF);
    float*    b2f = (float*)(blob + B2F_OFF);
    float*    b3f = (float*)(blob + B3F_OFF);

    // W1/b1 -> A-layout f16: flat j = (kc*32 + lane)*16 + i
    for (int j = t; j < 1024; j += 256) {
        int kc = j >> 9, lane = (j >> 4) & 31, i = j & 15;
        int feat = kc * 32 + kmapA(lane, i);
        w1s[j] = (_Float16)W1[s * HID + feat];
        b1s[j] = (_Float16)b1[s * HID + feat];
    }
    // W2 -> 8 B-operand tiles (tile = nt*2 + kc): lane holds col n, element i -> row k
    for (int j = t; j < 4096; j += 256) {
        int tile = j >> 9, lane = (j >> 4) & 31, i = j & 15;
        int nt = tile >> 1, kc = tile & 1;
        int n = nt * 16 + (lane & 15);
        int k = kc * 32 + kmapA(lane, i);
        w2s[j] = (_Float16)W2[(size_t)s * HID * HID + k * HID + n];
    }
    if (t < HID) { w3f[t] = W3[s * HID + t]; b2f[t] = b2[s * HID + t]; }
    if (t == 0)  { b3f[0] = b3[s]; }
}

// ---------------- main simulation kernel ----------------
__global__ __launch_bounds__(256)
void bsde_sim_kernel(const float* __restrict__ S0p, const float* __restrict__ noise,
                     const float* __restrict__ Y0p, const float* __restrict__ Z0p,
                     const uint8_t* __restrict__ ws, float* __restrict__ out)
{
    __shared__ alignas(16) uint8_t smem[2][STEP_BYTES]; // double-buffered weight blob
    __shared__ float zbuf[8][64];                       // per-wave Z scatter buffer

    const int t    = threadIdx.x;
    const int lane = t & 31;
    const int wave = t >> 5;
    const int gp0  = blockIdx.x * 512 + wave * 64 + lane;  // this lane's path 0
    const int gp1  = gp0 + 32;                             // this lane's path 1

    const float S0v = S0p[0];
    const float Z0v = Z0p[0];
    float X0 = S0v, X1 = S0v;
    float Yv0 = Y0p[0], Yv1 = Yv0;

    const unsigned long long wsbase = (unsigned long long)(uintptr_t)ws;

    // issue one 16KB blob as 4 async b128 loads per thread (ASYNCcnt += 4 per wave)
    auto issue_blob = [&](int buf, int blobIdx) {
        uint32_t ldsb = (uint32_t)(uintptr_t)(&smem[buf][0]) + (uint32_t)(t * 16);
        uint32_t gob  = (uint32_t)(blobIdx * STEP_BYTES + t * 16);
        #pragma unroll
        for (int i = 0; i < 4; ++i) {
            uint32_t lds = ldsb + (uint32_t)(i * 4096);
            uint32_t go  = gob  + (uint32_t)(i * 4096);
            asm volatile("global_load_async_to_lds_b128 %0, %1, %2"
                         :: "v"(lds), "v"(go), "s"(wsbase) : "memory");
        }
    };

    // prefetch weights for step 1 into buffer 1
    issue_blob(1, 0);

    // ---- step 0: constant Z0 ----
    {
        float dW0 = __builtin_nontemporal_load(&noise[gp0]) * SQRT_DT;
        float dW1 = __builtin_nontemporal_load(&noise[gp1]) * SQRT_DT;
        Yv0 = Yv0 - R_F * Yv0 * DT + Z0v * SIG * X0 * dW0;
        Yv1 = Yv1 - R_F * Yv1 * DT + Z0v * SIG * X1 * dW1;
        X0 *= __expf(DRIFT + SIG * dW0);
        X1 *= __expf(DRIFT + SIG * dW1);
    }

    for (int s = 1; s < NSTEPS; ++s) {
        const int cur = s & 1;
        // prefetch next step's weights into the other buffer (dummy re-load on last step)
        issue_blob(cur ^ 1, (s < NSTEPS - 1) ? s : (NSTEPS - 2));
        // current blob's 4 loads were issued first; async loads complete in order
        asm volatile("s_wait_asynccnt 0x4" ::: "memory");
        __syncthreads();

        const uint8_t* blob = smem[cur];

        // hoisted per-lane operand loads from LDS
        v16h w1v0 = *(const v16h*)(blob + W1S_OFF + (0 * 32 + lane) * 32);
        v16h w1v1 = *(const v16h*)(blob + W1S_OFF + (1 * 32 + lane) * 32);
        v16h b1v0 = *(const v16h*)(blob + B1S_OFF + (0 * 32 + lane) * 32);
        v16h b1v1 = *(const v16h*)(blob + B1S_OFF + (1 * 32 + lane) * 32);
        v16h bt[8];
        #pragma unroll
        for (int tt = 0; tt < 8; ++tt)
            bt[tt] = *(const v16h*)(blob + W2S_OFF + (tt * 32 + lane) * 32);
        const float* w3f = (const float*)(blob + W3F_OFF);
        const float* b2f = (const float*)(blob + B2F_OFF);
        float w3v[4], b2v[4];
        #pragma unroll
        for (int nt = 0; nt < 4; ++nt) {
            w3v[nt] = w3f[nt * 16 + (lane & 15)];
            b2v[nt] = b2f[nt * 16 + (lane & 15)];
        }
        const float b3v = *(const float*)(blob + B3F_OFF);

        const float xn0 = X0 / S0v;
        const float xn1 = X1 / S0v;

        // 4 M-tiles of 16 paths each (wave owns 64 paths)
        #pragma unroll
        for (int mt = 0; mt < 4; ++mt) {
            // x of row m = lane%16 of this tile
            float xm = __shfl((mt < 2) ? xn0 : xn1, ((mt & 1) << 4) + (lane & 15), 32);
            _Float16 xh = (_Float16)xm;

            // layer 1 (rank-1) straight into A-operand layout, f16 (pk_fma/pk_max)
            v16h a0, a1;
            #pragma unroll
            for (int i = 0; i < 16; ++i) {
                _Float16 h0 = xh * w1v0[i] + b1v0[i];
                _Float16 h1 = xh * w1v1[i] + b1v1[i];
                a0[i] = h0 > (_Float16)0 ? h0 : (_Float16)0;
                a1[i] = h1 > (_Float16)0 ? h1 : (_Float16)0;
            }

            // layer 2: [16x64] @ [64x64], bias fused into accumulator
            v8f c[4];
            #pragma unroll
            for (int nt = 0; nt < 4; ++nt) {
                #pragma unroll
                for (int r = 0; r < 8; ++r) c[nt][r] = b2v[nt];
                c[nt] = __builtin_amdgcn_wmma_f32_16x16x32_f16(
                    false, a0, false, bt[nt * 2 + 0], (short)0, c[nt], false, false);
                c[nt] = __builtin_amdgcn_wmma_f32_16x16x32_f16(
                    false, a1, false, bt[nt * 2 + 1], (short)0, c[nt], false, false);
            }

            // relu (single v_max_num_f32) + layer 3 partial dot
            float part[8];
            #pragma unroll
            for (int r = 0; r < 8; ++r) {
                float h0 = relu1(c[0][r]);
                float h1 = relu1(c[1][r]);
                float h2 = relu1(c[2][r]);
                float h3 = relu1(c[3][r]);
                part[r] = h0 * w3v[0] + h1 * w3v[1] + h2 * w3v[2] + h3 * w3v[3];
            }
            // butterfly sum over the 16 lanes sharing each row group
            #pragma unroll
            for (int m = 1; m <= 8; m <<= 1) {
                #pragma unroll
                for (int r = 0; r < 8; ++r)
                    part[r] += __shfl_xor(part[r], m, 32);
            }
            // lanes 0 and 16 own rows 0-7 / 8-15 of the tile: scatter Z
            if ((lane & 15) == 0) {
                int mbase = mt * 16 + ((lane >> 4) << 3);
                #pragma unroll
                for (int r = 0; r < 8; ++r)
                    zbuf[wave][mbase + r] = part[r] + b3v;
            }
        }
        __builtin_amdgcn_wave_barrier();   // order zbuf DS writes before reads (same wave)
        float Zl0 = zbuf[wave][lane];
        float Zl1 = zbuf[wave][lane + 32];

        // ---- SDE update (noise streamed non-temporally: read exactly once) ----
        size_t nb = (size_t)s * BATCH;
        float dW0 = __builtin_nontemporal_load(&noise[nb + gp0]) * SQRT_DT;
        float dW1 = __builtin_nontemporal_load(&noise[nb + gp1]) * SQRT_DT;
        Yv0 = Yv0 - R_F * Yv0 * DT + Zl0 * SIG * X0 * dW0;
        Yv1 = Yv1 - R_F * Yv1 * DT + Zl1 * SIG * X1 * dW1;
        X0 *= __expf(DRIFT + SIG * dW0);
        X1 *= __expf(DRIFT + SIG * dW1);

        __syncthreads();   // all waves done reading cur buffer before next prefetch hits it
    }

    const float disc = __expf(-R_F * T_F);
    __builtin_nontemporal_store(Yv0, &out[gp0]);
    __builtin_nontemporal_store(Yv1, &out[gp1]);
    __builtin_nontemporal_store(disc * fmaxf(X0 - KSTR, 0.0f), &out[BATCH + gp0]);
    __builtin_nontemporal_store(disc * fmaxf(X1 - KSTR, 0.0f), &out[BATCH + gp1]);
}

// ---------------- launch ----------------
extern "C" void kernel_launch(void* const* d_in, const int* in_sizes, int n_in,
                              void* d_out, int out_size, void* d_ws, size_t ws_size,
                              hipStream_t stream) {
    // input order: S0_val, batch_size, noise, Y0, Z0, W1, b1, W2, b2, W3, b3
    const float* S0p   = (const float*)d_in[0];
    const float* noise = (const float*)d_in[2];
    const float* Y0p   = (const float*)d_in[3];
    const float* Z0p   = (const float*)d_in[4];
    const float* W1    = (const float*)d_in[5];
    const float* b1    = (const float*)d_in[6];
    const float* W2    = (const float*)d_in[7];
    const float* b2    = (const float*)d_in[8];
    const float* W3    = (const float*)d_in[9];
    const float* b3    = (const float*)d_in[10];
    uint8_t* ws = (uint8_t*)d_ws;
    float* out = (float*)d_out;

    bsde_prep_kernel<<<NSTEPS - 1, 256, 0, stream>>>(W1, b1, W2, b2, W3, b3, ws);
    bsde_sim_kernel<<<BATCH / 512, 256, 0, stream>>>(S0p, noise, Y0p, Z0p, ws, out);
    (void)in_sizes; (void)n_in; (void)out_size; (void)ws_size;
}
